// PointNeXtSetAbstraction_70772471103817
// MI455X (gfx1250) — matrix operand order repaired and checked by reading, not compiled, not run on hardware
//
#include <hip/hip_runtime.h>
#include <hip/hip_bf16.h>

// ---------------- problem constants ----------------
#define BB   8
#define NN   8192
#define DD   3
#define CC   64
#define MM   2048
#define KK   32
#define R2   0.09f          // RADIUS^2
#define OUTD 128
#define CIN  67
#define HID  268
#define EPS  1e-6f

// padded GEMM dims
#define CINP  96            // K-dim pad for GEMM1 (3 x 32)
#define HIDP  272           // N pad GEMM1 (17 x 16)
#define HIDP2 288           // K pad GEMM2 (9 x 32)

typedef __attribute__((ext_vector_type(16))) __bf16 v16bf;
typedef __attribute__((ext_vector_type(8)))  float  v8f;

// ---------------- helpers ----------------
__device__ __forceinline__ float tanh_fast(float u) {
#if __has_builtin(__builtin_amdgcn_tanhf)
    return __builtin_amdgcn_tanhf(u);                 // v_tanh_f32 (gfx1250 TRANS op)
#elif __has_builtin(__builtin_amdgcn_tanh_f32)
    return __builtin_amdgcn_tanh_f32(u);
#else
    // branch-free: tanh(u) = sign(u) * (1 - e) / (1 + e), e = exp(-2|u|)  (v_exp_f32 path)
    float a = __builtin_fabsf(u);
    float e = __expf(-2.0f * a);
    float t = (1.0f - e) / (1.0f + e);
    return __builtin_copysignf(t, u);
#endif
}

__device__ __forceinline__ float gelu_tanh(float x) {
    float x3 = x * x * x;
    return 0.5f * x * (1.0f + tanh_fast(0.7978845608028654f * (x + 0.044715f * x3)));
}

// A fragment: 16x32 bf16, row-major source with leading dim ld.
// ISA 16-bit A layout: lanes 0-15 hold row M=lane, elems 0-7 -> K 0..7, elems 8-15 -> K 16..23;
// lanes 16-31 same rows, K 8..15 and K 24..31.
__device__ __forceinline__ v16bf load_a_frag(const __bf16* base, int ld, int rbase, int kbase, int lane) {
    int row  = rbase + (lane & 15);
    int half = lane >> 4;
    const __bf16* p = base + row * ld + kbase + half * 8;
    v16bf a;
#pragma unroll
    for (int i = 0; i < 8; ++i) { a[i] = p[i]; a[i + 8] = p[i + 16]; }
    return a;
}

// B fragment: 32x16 bf16 from column(n)-major storage (k contiguous), leading dim ld (K extent).
// lanes 0-15: column N=lane, K kbase+0..7 and kbase+16..23 ; lanes 16-31: column N=lane-16, K kbase+8..15 and kbase+24..31.
__device__ __forceinline__ v16bf load_b_frag(const __bf16* baseT, int ld, int nbase, int kbase, int lane) {
    int col  = nbase + (lane & 15);
    int half = lane >> 4;
    const __bf16* p = baseT + col * ld + kbase + half * 8;
    v16bf b;
#pragma unroll
    for (int i = 0; i < 8; ++i) { b[i] = p[i]; b[i + 8] = p[i + 16]; }
    return b;
}

// ---------------- kernel 0: weight convert / pad / transpose to bf16 ----------------
__global__ void cvt_w1_kernel(const float* __restrict__ w1, __bf16* __restrict__ w1bT) {
    int i = blockIdx.x * blockDim.x + threadIdx.x;     // over HIDP * CINP
    if (i >= HIDP * CINP) return;
    int n = i / CINP, k = i % CINP;
    float v = (n < HID && k < CIN) ? w1[k * HID + n] : 0.0f;
    w1bT[i] = (__bf16)v;
}

__global__ void cvt_w2_kernel(const float* __restrict__ w2, __bf16* __restrict__ w2bT) {
    int i = blockIdx.x * blockDim.x + threadIdx.x;     // over OUTD * HIDP2
    if (i >= OUTD * HIDP2) return;
    int n = i / HIDP2, k = i % HIDP2;
    float v = (k < HID) ? w2[k * OUTD + n] : 0.0f;
    w2bT[i] = (__bf16)v;
}

// ---------------- kernel 1: farthest point sampling (one block per batch) ----------------
__global__ __launch_bounds__(512) void fps_kernel(const float* __restrict__ x,
                                                  const int* __restrict__ first_idx,
                                                  float* __restrict__ centers) {
    __shared__ float sdist[NN];          // 32 KB
    __shared__ float sval[512];
    __shared__ int   sidx[512];
    __shared__ float sbc[3];

    const int b   = blockIdx.x;
    const int tid = threadIdx.x;
    const float* xb = x + (size_t)b * NN * DD;

    int f0 = first_idx[b];
    float fx = xb[f0 * 3 + 0], fy = xb[f0 * 3 + 1], fz = xb[f0 * 3 + 2];
    for (int n = tid; n < NN; n += 512) {
        float dx = xb[n * 3 + 0] - fx, dy = xb[n * 3 + 1] - fy, dz = xb[n * 3 + 2] - fz;
        sdist[n] = dx * dx + dy * dy + dz * dz;
    }
    if (tid == 0) {
        centers[((size_t)b * MM + 0) * 3 + 0] = fx;
        centers[((size_t)b * MM + 0) * 3 + 1] = fy;
        centers[((size_t)b * MM + 0) * 3 + 2] = fz;
    }
    __syncthreads();

    for (int i = 1; i < MM; ++i) {
        // local argmax (first occurrence wins on ties -> strict > while scanning ascending)
        float bv = -1.0f; int bi = 0;
        for (int n = tid; n < NN; n += 512)
            if (sdist[n] > bv) { bv = sdist[n]; bi = n; }
        sval[tid] = bv; sidx[tid] = bi;
        __syncthreads();
        for (int s = 256; s > 0; s >>= 1) {
            if (tid < s) {
                float v2 = sval[tid + s]; int i2 = sidx[tid + s];
                if (v2 > sval[tid] || (v2 == sval[tid] && i2 < sidx[tid])) { sval[tid] = v2; sidx[tid] = i2; }
            }
            __syncthreads();
        }
        if (tid == 0) {
            int far = sidx[0];
            float px = xb[far * 3 + 0], py = xb[far * 3 + 1], pz = xb[far * 3 + 2];
            sbc[0] = px; sbc[1] = py; sbc[2] = pz;
            centers[((size_t)b * MM + i) * 3 + 0] = px;
            centers[((size_t)b * MM + i) * 3 + 1] = py;
            centers[((size_t)b * MM + i) * 3 + 2] = pz;
        }
        __syncthreads();
        float px = sbc[0], py = sbc[1], pz = sbc[2];
        for (int n = tid; n < NN; n += 512) {
            float dx = xb[n * 3 + 0] - px, dy = xb[n * 3 + 1] - py, dz = xb[n * 3 + 2] - pz;
            float d2 = dx * dx + dy * dy + dz * dz;
            sdist[n] = fminf(sdist[n], d2);
        }
        __syncthreads();
    }
}

// ---------------- kernel 2: ball query, one wave (32 lanes) per center ----------------
// Selects K nearest points with d2 < R2, sorted ascending by (d2, idx); empty slots get N-1
// (reference stores -1, which JAX index-wraps to N-1).
__global__ __launch_bounds__(256) void ball_kernel(const float* __restrict__ x,
                                                   const float* __restrict__ centers,
                                                   int* __restrict__ nbr) {
    const int lane = threadIdx.x & 31;
    const int wave = threadIdx.x >> 5;
    const int g    = blockIdx.x * 8 + wave;       // center id
    const int b    = g / MM;

    const float cx = centers[(size_t)g * 3 + 0];
    const float cy = centers[(size_t)g * 3 + 1];
    const float cz = centers[(size_t)g * 3 + 2];
    const float* xb = x + (size_t)b * NN * DD;
    int* nb = nbr + (size_t)g * KK;

    const unsigned long long SENT = ~0ull;
    unsigned long long prev = 0ull;
    for (int r = 0; r < KK; ++r) {
        // each lane scans its strided points for min lexicographic key strictly greater than prev
        unsigned long long best = SENT;
        for (int n = lane; n < NN; n += 32) {
            float dx = xb[n * 3 + 0] - cx, dy = xb[n * 3 + 1] - cy, dz = xb[n * 3 + 2] - cz;
            float d2 = dx * dx + dy * dy + dz * dz;
            if (d2 < R2) {
                unsigned int bits = __float_as_uint(d2);      // d2 >= 0 -> monotonic bits
                unsigned long long key = ((unsigned long long)bits << 32) | (unsigned int)(n + 1);
                if (key > prev && key < best) best = key;
            }
        }
        // wave32 min-reduce
        for (int off = 16; off > 0; off >>= 1) {
            unsigned long long o = __shfl_xor(best, off, 32);
            if (o < best) best = o;
        }
        if (best == SENT) {
            if (lane == 0)
                for (int rr = r; rr < KK; ++rr) nb[rr] = NN - 1;
            break;
        }
        if (lane == 0) nb[r] = (int)((best & 0xffffffffu) - 1u);
        prev = best;
    }
}

// ---------------- kernel 3: fused MLP (WMMA bf16) + layernorms + max-pool ----------------
__global__ __launch_bounds__(128) void mlp_kernel(const float* __restrict__ x,
                                                  const float* __restrict__ features,
                                                  const float* __restrict__ centers,
                                                  const int* __restrict__ nbr,
                                                  const __bf16* __restrict__ w1bT,
                                                  const float* __restrict__ b1,
                                                  const float* __restrict__ g1,
                                                  const float* __restrict__ be1,
                                                  const __bf16* __restrict__ w2bT,
                                                  const float* __restrict__ b2,
                                                  const float* __restrict__ g2,
                                                  const float* __restrict__ be2,
                                                  float* __restrict__ outp) {
    __shared__ __bf16 sComb[KK * CINP];    //  6 KB : gathered activations (bf16, zero-padded K)
    __shared__ float  sH1[KK * HIDP];      // 34 KB : gelu(GEMM1) fp32
    __shared__ __bf16 sA2[KK * HIDP2];     // 18 KB : LN1 output, bf16, zero-padded K
    __shared__ float  sH2[KK * OUTD];      // 16 KB : GEMM2 fp32
    __shared__ float  sMu[KK], sRs[KK];

    const int tid  = threadIdx.x;
    const int lane = tid & 31;
    const int wave = tid >> 5;
    const int bm   = blockIdx.x;           // center id
    const int b    = bm / MM;

    const float cx = centers[(size_t)bm * 3 + 0];
    const float cy = centers[(size_t)bm * 3 + 1];
    const float cz = centers[(size_t)bm * 3 + 2];
    const int* nb = nbr + (size_t)bm * KK;

    // ---- gather comb = [features | rel] into LDS as bf16, K-padded with zeros ----
    for (int e = tid; e < KK * CINP; e += 128) {
        int r = e / CINP, c = e % CINP;
        int idx = nb[r];
        float v = 0.0f;
        if (c < CC) {
            v = features[((size_t)b * NN + idx) * CC + c];
        } else if (c < CIN) {
            int d = c - CC;
            float ctr = (d == 0) ? cx : (d == 1) ? cy : cz;
            v = x[((size_t)b * NN + idx) * 3 + d] - ctr;
        }
        sComb[e] = (__bf16)v;
    }
    __syncthreads();

    // ---- GEMM1: [32 x 96] x [96 x 272] -> gelu -> sH1 ----
    // mtile outer so the 3 A-fragments are loaded from LDS once per mtile (LICM-hoisted).
    const int col  = lane & 15;
    const int half = lane >> 4;
#pragma unroll
    for (int mt = 0; mt < 2; ++mt) {
        const int rbase = mt * 16;
        v16bf a0 = load_a_frag(sComb, CINP, rbase,  0, lane);
        v16bf a1 = load_a_frag(sComb, CINP, rbase, 32, lane);
        v16bf a2 = load_a_frag(sComb, CINP, rbase, 64, lane);
        for (int nt = wave; nt < 17; nt += 4) {
            const int nbase = nt * 16;
            v8f acc = {};
            acc = __builtin_amdgcn_wmma_f32_16x16x32_bf16(false, a0, false,
                      load_b_frag(w1bT, CINP, nbase,  0, lane), (short)0, acc, false, false);
            acc = __builtin_amdgcn_wmma_f32_16x16x32_bf16(false, a1, false,
                      load_b_frag(w1bT, CINP, nbase, 32, lane), (short)0, acc, false, false);
            acc = __builtin_amdgcn_wmma_f32_16x16x32_bf16(false, a2, false,
                      load_b_frag(w1bT, CINP, nbase, 64, lane), (short)0, acc, false, false);
            int n = nbase + col;
            // unconditional clamped load + select: no EXEC divergence around the load
            float bias = b1[n < HID ? n : HID - 1];
            bias = (n < HID) ? bias : 0.0f;
#pragma unroll
            for (int j = 0; j < 8; ++j) {
                int row = rbase + j + 8 * half;
                sH1[row * HIDP + n] = gelu_tanh(acc[j] + bias);
            }
        }
    }
    __syncthreads();

    // ---- LayerNorm1 over HID, write bf16 (K-padded) for GEMM2 ----
    if (tid < KK) {
        int r = tid;
        float s = 0.0f, sq = 0.0f;
        for (int n = 0; n < HID; ++n) { float h = sH1[r * HIDP + n]; s += h; sq += h * h; }
        float mu = s / (float)HID;
        float rs = rsqrtf(sq / (float)HID - mu * mu + EPS);
        for (int n = 0; n < HIDP2; ++n) {
            float v = 0.0f;
            if (n < HID) v = (sH1[r * HIDP + n] - mu) * rs * g1[n] + be1[n];
            sA2[r * HIDP2 + n] = (__bf16)v;
        }
    }
    __syncthreads();

    // ---- GEMM2: [32 x 288] x [288 x 128] -> sH2 ; A-fragments hoisted per mtile ----
#pragma unroll
    for (int mt = 0; mt < 2; ++mt) {
        const int rbase = mt * 16;
        v16bf af[9];
#pragma unroll
        for (int ks = 0; ks < 9; ++ks) af[ks] = load_a_frag(sA2, HIDP2, rbase, ks * 32, lane);
        for (int nt = wave; nt < 8; nt += 4) {
            const int nbase = nt * 16;
            v8f acc = {};
#pragma unroll
            for (int ks = 0; ks < 9; ++ks) {
                acc = __builtin_amdgcn_wmma_f32_16x16x32_bf16(false, af[ks], false,
                          load_b_frag(w2bT, HIDP2, nbase, ks * 32, lane), (short)0, acc, false, false);
            }
            int n = nbase + col;
            float bias = b2[n];
#pragma unroll
            for (int j = 0; j < 8; ++j) {
                int row = rbase + j + 8 * half;
                sH2[row * OUTD + n] = acc[j] + bias;
            }
        }
    }
    __syncthreads();

    // ---- LayerNorm2 stats per row ----
    if (tid < KK) {
        int r = tid;
        float s = 0.0f, sq = 0.0f;
        for (int n = 0; n < OUTD; ++n) { float h = sH2[r * OUTD + n]; s += h; sq += h * h; }
        float mu = s / (float)OUTD;
        sMu[r] = mu;
        sRs[r] = rsqrtf(sq / (float)OUTD - mu * mu + EPS);
    }
    __syncthreads();

    // ---- affine + max-pool over the K neighbors, write output ----
    {
        int n = tid;            // 128 threads == OUTD
        float gn = g2[n], bn = be2[n];
        float mx = -3.402823466e+38f;
        for (int r = 0; r < KK; ++r) {
            float v = (sH2[r * OUTD + n] - sMu[r]) * sRs[r] * gn + bn;
            mx = fmaxf(mx, v);
        }
        outp[(size_t)bm * OUTD + n] = mx;
    }
}

// ---------------- launcher ----------------
extern "C" void kernel_launch(void* const* d_in, const int* in_sizes, int n_in,
                              void* d_out, int out_size, void* d_ws, size_t ws_size,
                              hipStream_t stream) {
    const float* x         = (const float*)d_in[0];
    const float* features  = (const float*)d_in[1];
    const int*   first_idx = (const int*)d_in[2];
    const float* w1        = (const float*)d_in[3];
    const float* b1        = (const float*)d_in[4];
    const float* g1        = (const float*)d_in[5];
    const float* be1       = (const float*)d_in[6];
    const float* w2        = (const float*)d_in[7];
    const float* b2        = (const float*)d_in[8];
    const float* g2        = (const float*)d_in[9];
    const float* be2       = (const float*)d_in[10];

    float* centers = (float*)d_out;                         // B*M*3 floats
    float* outp    = (float*)d_out + (size_t)BB * MM * 3;   // B*M*OUT floats

    // workspace layout (256B aligned segments)
    char* ws = (char*)d_ws;
    int*    nbr  = (int*)ws;                                        // B*M*K ints = 2 MB
    size_t  off  = (size_t)BB * MM * KK * sizeof(int);
    __bf16* w1bT = (__bf16*)(ws + off);                             // HIDP*CINP bf16
    off += (size_t)HIDP * CINP * sizeof(__bf16);
    off  = (off + 255) & ~(size_t)255;
    __bf16* w2bT = (__bf16*)(ws + off);                             // OUTD*HIDP2 bf16

    // 0) weight convert/pad/transpose
    cvt_w1_kernel<<<(HIDP * CINP + 255) / 256, 256, 0, stream>>>(w1, w1bT);
    cvt_w2_kernel<<<(OUTD * HIDP2 + 255) / 256, 256, 0, stream>>>(w2, w2bT);

    // 1) farthest point sampling -> centers (into d_out directly)
    fps_kernel<<<BB, 512, 0, stream>>>(x, first_idx, centers);

    // 2) ball query -> nbr (one wave per center, 8 waves per block)
    ball_kernel<<<(BB * MM) / 8, 256, 0, stream>>>(x, centers, nbr);

    // 3) fused gather + MLP (WMMA bf16) + LN + max-pool -> out
    mlp_kernel<<<BB * MM, 128, 0, stream>>>(x, features, centers, nbr,
                                            w1bT, b1, g1, be1,
                                            w2bT, b2, g2, be2, outp);
}
